// IntraMetapathAggregation_67362267071041
// MI455X (gfx1250) — matrix operand order repaired and checked by reading, not compiled, not run on hardware
//
#include <hip/hip_runtime.h>
#include <hip/hip_bf16.h>
#include <math.h>

// ---------------------------------------------------------------------------
// MI455X (gfx1250) implementation of MAGNN intra-metapath aggregation.
// Phases:
//   0) init:   out=0, seg_max=-inf, seg_sum=0
//   1) prep_w: W_enc (128x512 f32, row-major) -> col-major bf16 copy in ws
//   2) er:     er[n][k] = node_feat[n] . W_r[k]
//   3) logits: persistent blocks, B (W_enc) register-resident; per 16-edge
//              tile: mean over L -> bf16 LDS, WMMA GEMM, el = <ft, attn_l>,
//              e = leakyrelu(el + er[dst]), atomic max into seg_max
//   4) exp_sum: ex = exp(e - seg_max[dst]), atomic add into seg_sum
//   5) recip:  seg_sum -> 1/seg_sum
//   6) scatter: recompute ft via WMMA (B still register-resident),
//              out[dst] += ft * ex * (1/seg_sum)
// GEMM is recomputed in phase 6 instead of storing ft (655MB) to HBM:
// re-reading edge_feat (491MB) + WMMA is cheaper at 23.3 TB/s.
// Weight-stationary: each wave owns one head's 64 columns; its 16 B
// fragments (128 VGPRs) are loaded once per persistent block and reused
// across all edge tiles the block processes.
// ---------------------------------------------------------------------------

typedef __attribute__((ext_vector_type(16))) __bf16 v16bf;
typedef __attribute__((ext_vector_type(8)))  __bf16 v8bf;
typedef __attribute__((ext_vector_type(8)))  float  v8f;

#define KH     8      // heads
#define HD     64     // dim per head
#define DIN    128    // input feature dim
#define LEN    3      // metapath length
#define TILE_E 16     // edges per tile (WMMA M)
#define NEG_SLOPE 0.01f
#define LDS_PAD 8     // pad LDS rows: stride 136*2B = 68 dwords -> conflict-free
#define PERSIST_BLOCKS 2048

__device__ __forceinline__ void atomic_max_f32(float* addr, float val) {
    // sign-aware integer punning: correct for IEEE754 with -inf init
    if (val >= 0.0f)
        atomicMax((int*)addr, __float_as_int(val));
    else
        atomicMin((unsigned int*)addr, (unsigned int)__float_as_int(val));
}

// ---------------- phase 0: init output / segment buffers --------------------
__global__ void init_kernel(float* __restrict__ out,
                            float* __restrict__ seg_max,
                            float* __restrict__ seg_sum, int n_nodes) {
    int i = blockIdx.x * blockDim.x + threadIdx.x;
    if (i < n_nodes * KH * HD) out[i] = 0.0f;
    if (i < n_nodes * KH) { seg_max[i] = -INFINITY; seg_sum[i] = 0.0f; }
}

// ---------------- phase 1: W_enc -> col-major bf16 --------------------------
__global__ void prep_w_kernel(const float* __restrict__ W_enc,
                              __bf16* __restrict__ W_bf) {
    int i = blockIdx.x * blockDim.x + threadIdx.x;   // i = n*128 + k
    if (i >= DIN * KH * HD) return;
    int n = i >> 7, k = i & 127;
    W_bf[n * DIN + k] = (__bf16)W_enc[k * (KH * HD) + n];
}

// ---------------- phase 2: er = node_feat @ W_r.T ---------------------------
__global__ void er_kernel(const float* __restrict__ node_feat,
                          const float* __restrict__ W_r,
                          float* __restrict__ er, int n_nodes) {
    int wave = threadIdx.x >> 5;
    int lane = threadIdx.x & 31;
    int n = blockIdx.x * 8 + wave;
    if (n >= n_nodes) return;
    const float4 nf = ((const float4*)(node_feat + (size_t)n * DIN))[lane];
    for (int k = 0; k < KH; ++k) {
        const float4 wr = ((const float4*)(W_r + k * DIN))[lane];
        float d = nf.x * wr.x + nf.y * wr.y + nf.z * wr.z + nf.w * wr.w;
        #pragma unroll
        for (int m = 16; m >= 1; m >>= 1) d += __shfl_xor(d, m, 32);
        if (lane == 0) er[n * KH + k] = d;
    }
}

// ---- load this wave's 16 B fragments (head's 64 cols x 128 K) to VGPRs ----
__device__ __forceinline__ void load_b_frags(const __bf16* __restrict__ W_bf,
                                             int wave, int lane,
                                             v16bf B[4][4]) {
    int hi = lane >> 4, nl = lane & 15;
    #pragma unroll
    for (int ki = 0; ki < 4; ++ki) {
        int kk = ki * 32;
        #pragma unroll
        for (int t = 0; t < 4; ++t) {
            int col = wave * HD + t * 16 + nl;       // output column 0..511
            B[ki][t] = *(const v16bf*)(W_bf + (size_t)col * DIN + kk + 16 * hi);
        }
    }
}

// ---- stage mean(edge_feat) tile (16x128) as bf16 into padded LDS ----------
__device__ __forceinline__ void stage_a_tile(
        const float* __restrict__ edge_feat, int e0, int e_cnt,
        __bf16 (*Asm)[DIN + LDS_PAD]) {
    for (int i = threadIdx.x; i < TILE_E * DIN; i += blockDim.x) {
        int r = i >> 7, d = i & 127;
        int e = e0 + r; if (e >= e_cnt) e = e_cnt - 1;   // clamp (E%16==0 anyway)
        const float* p = edge_feat + (size_t)e * (LEN * DIN) + d;
        float s = (p[0] + p[DIN] + p[2 * DIN]) * (1.0f / 3.0f);
        Asm[r][d] = (__bf16)s;
    }
    // prefetch next tile's edge features while we compute
    __builtin_prefetch(edge_feat + (size_t)(e0 + TILE_E) * (LEN * DIN)
                       + (threadIdx.x << 3), 0, 0);
}

// ---- 16 WMMAs: A (LDS) x B (registers) -> 16x64 f32 tile per wave ---------
__device__ __forceinline__ void gemm_tile(const __bf16 (*Asm)[DIN + LDS_PAD],
                                          const v16bf B[4][4],
                                          int lane, v8f acc[4]) {
    int hi = lane >> 4, nl = lane & 15;
    v8f zero = {0,0,0,0,0,0,0,0};
    acc[0] = zero; acc[1] = zero; acc[2] = zero; acc[3] = zero;
    #pragma unroll
    for (int ki = 0; ki < 4; ++ki) {
        int kk = ki * 32;
        // A fragment, contiguous k (same as B): lanes0-15 K[kk..+15];
        //                            lanes16-31 K[kk+16..+31]
        union { v16bf v; v8bf h[2]; } a;
        a.h[0] = *(const v8bf*)&Asm[nl][kk + 16 * hi];
        a.h[1] = *(const v8bf*)&Asm[nl][kk + 16 * hi + 8];
        #pragma unroll
        for (int t = 0; t < 4; ++t) {
            acc[t] = __builtin_amdgcn_wmma_f32_16x16x32_bf16(
                false, a.v, false, B[ki][t], (short)0, acc[t], false, false);
        }
    }
}

// ---------------- phase 3: attention logits + segment max -------------------
__global__ void __launch_bounds__(256)
logits_kernel(const float* __restrict__ edge_feat,
              const __bf16* __restrict__ W_bf,
              const float* __restrict__ attn_l,
              const float* __restrict__ er,
              const int* __restrict__ dst,
              float* __restrict__ e_ws,
              float* __restrict__ seg_max, int e_cnt) {
    __shared__ __bf16 Asm[TILE_E][DIN + LDS_PAD];
    int wave = threadIdx.x >> 5, lane = threadIdx.x & 31;
    int hi = lane >> 4, nl = lane & 15;

    v16bf B[4][4];
    load_b_frags(W_bf, wave, lane, B);          // weight-stationary
    float attnw[4];
    #pragma unroll
    for (int t = 0; t < 4; ++t) attnw[t] = attn_l[wave * HD + t * 16 + nl];

    int ntiles = (e_cnt + TILE_E - 1) / TILE_E;
    for (int tile = blockIdx.x; tile < ntiles; tile += gridDim.x) {
        int e0 = tile * TILE_E;
        stage_a_tile(edge_feat, e0, e_cnt, Asm);
        __syncthreads();
        v8f acc[4];
        gemm_tile(Asm, B, lane, acc);

        // el[row] = sum_h ft[row][h] * attn_l[head][h]; lanes hold disjoint h
        float partial[8];
        #pragma unroll
        for (int r = 0; r < 8; ++r) partial[r] = 0.0f;
        #pragma unroll
        for (int t = 0; t < 4; ++t)
            #pragma unroll
            for (int r = 0; r < 8; ++r) partial[r] += acc[t][r] * attnw[t];
        // reduce across the 16 lanes of each half-wave (rows r+8*hi)
        #pragma unroll
        for (int m = 1; m <= 8; m <<= 1) {
            #pragma unroll
            for (int r = 0; r < 8; ++r)
                partial[r] += __shfl_xor(partial[r], m, 32);
        }
        if (nl == 0) {  // lane 0 -> rows 0..7, lane 16 -> rows 8..15
            #pragma unroll
            for (int r = 0; r < 8; ++r) {
                int e = e0 + r + 8 * hi;
                if (e < e_cnt) {
                    int d = dst[e];
                    float x = partial[r] + er[d * KH + wave];
                    x = (x > 0.0f) ? x : NEG_SLOPE * x;       // leaky relu
                    e_ws[(size_t)e * KH + wave] = x;
                    atomic_max_f32(&seg_max[d * KH + wave], x);
                }
            }
        }
        __syncthreads();   // WAR: Asm reused next iteration
    }
}

// ---------------- phase 4: ex = exp(e - max), segment sum -------------------
__global__ void exp_sum_kernel(const int* __restrict__ dst,
                               const float* __restrict__ seg_max,
                               float* __restrict__ e_ws,   // in: e, out: ex
                               float* __restrict__ seg_sum, int e_cnt) {
    int i = blockIdx.x * blockDim.x + threadIdx.x;
    if (i >= e_cnt * KH) return;
    int e = i >> 3, k = i & 7;
    int d = dst[e];
    float ex = __expf(e_ws[i] - seg_max[d * KH + k]);
    e_ws[i] = ex;
    atomicAdd(&seg_sum[d * KH + k], ex);
}

// ---------------- phase 5: seg_sum -> reciprocal ----------------------------
__global__ void recip_kernel(float* __restrict__ seg_sum, int n) {
    int i = blockIdx.x * blockDim.x + threadIdx.x;
    if (i < n) seg_sum[i] = 1.0f / seg_sum[i];
}

// ---------------- phase 6: recompute ft, weighted scatter -------------------
__global__ void __launch_bounds__(256)
scatter_kernel(const float* __restrict__ edge_feat,
               const __bf16* __restrict__ W_bf,
               const float* __restrict__ ex_ws,
               const float* __restrict__ rsum,   // 1/seg_sum
               const int* __restrict__ dst,
               float* __restrict__ out, int e_cnt) {
    __shared__ __bf16 Asm[TILE_E][DIN + LDS_PAD];
    int wave = threadIdx.x >> 5, lane = threadIdx.x & 31;
    int hi = lane >> 4, nl = lane & 15;

    v16bf B[4][4];
    load_b_frags(W_bf, wave, lane, B);          // weight-stationary

    int ntiles = (e_cnt + TILE_E - 1) / TILE_E;
    for (int tile = blockIdx.x; tile < ntiles; tile += gridDim.x) {
        int e0 = tile * TILE_E;
        stage_a_tile(edge_feat, e0, e_cnt, Asm);
        __syncthreads();
        v8f acc[4];
        gemm_tile(Asm, B, lane, acc);

        float a_r[8]; int dst_r[8]; bool valid[8];
        #pragma unroll
        for (int r = 0; r < 8; ++r) {
            int e = e0 + r + 8 * hi;
            valid[r] = (e < e_cnt);
            int ec = valid[r] ? e : (e_cnt - 1);
            int d = dst[ec];
            dst_r[r] = d;
            a_r[r] = ex_ws[(size_t)ec * KH + wave] * rsum[d * KH + wave];
        }
        #pragma unroll
        for (int t = 0; t < 4; ++t) {
            int col = wave * HD + t * 16 + nl;
            #pragma unroll
            for (int r = 0; r < 8; ++r) {
                if (valid[r])
                    atomicAdd(&out[(size_t)dst_r[r] * (KH * HD) + col],
                              acc[t][r] * a_r[r]);
            }
        }
        __syncthreads();   // WAR: Asm reused next iteration
    }
}

// ---------------------------------------------------------------------------
extern "C" void kernel_launch(void* const* d_in, const int* in_sizes, int n_in,
                              void* d_out, int out_size, void* d_ws, size_t ws_size,
                              hipStream_t stream) {
    const float* node_feat = (const float*)d_in[0];
    const float* edge_feat = (const float*)d_in[1];
    const float* W_enc     = (const float*)d_in[2];
    const float* attn_l    = (const float*)d_in[3];
    const float* W_r       = (const float*)d_in[4];
    const int*   dst       = (const int*)d_in[5];

    const int n_nodes = in_sizes[0] / DIN;   // 20000
    const int e_cnt   = in_sizes[5];         // 320000
    float* out = (float*)d_out;

    // workspace carve-out (~12.3 MB total)
    char* ws = (char*)d_ws;
    size_t off = 0;
    auto carve = [&](size_t bytes) { void* p = ws + off;
                                     off = (off + bytes + 255) & ~(size_t)255;
                                     return p; };
    __bf16* W_bf   = (__bf16*)carve((size_t)KH * HD * DIN * sizeof(__bf16));
    float*  er     = (float*)carve((size_t)n_nodes * KH * sizeof(float));
    float*  seg_mx = (float*)carve((size_t)n_nodes * KH * sizeof(float));
    float*  seg_sm = (float*)carve((size_t)n_nodes * KH * sizeof(float));
    float*  e_ws   = (float*)carve((size_t)e_cnt * KH * sizeof(float));
    (void)ws_size; (void)n_in; (void)out_size;

    const int ntiles = (e_cnt + TILE_E - 1) / TILE_E;      // 20000
    const int gblk   = ntiles < PERSIST_BLOCKS ? ntiles : PERSIST_BLOCKS;

    init_kernel<<<(n_nodes * KH * HD + 255) / 256, 256, 0, stream>>>(
        out, seg_mx, seg_sm, n_nodes);
    prep_w_kernel<<<(DIN * KH * HD + 255) / 256, 256, 0, stream>>>(W_enc, W_bf);
    er_kernel<<<(n_nodes + 7) / 8, 256, 0, stream>>>(node_feat, W_r, er, n_nodes);
    logits_kernel<<<gblk, 256, 0, stream>>>(edge_feat, W_bf, attn_l, er, dst,
                                            e_ws, seg_mx, e_cnt);
    exp_sum_kernel<<<(e_cnt * KH + 255) / 256, 256, 0, stream>>>(
        dst, seg_mx, e_ws, seg_sm, e_cnt);
    recip_kernel<<<(n_nodes * KH + 255) / 256, 256, 0, stream>>>(
        seg_sm, n_nodes * KH);
    scatter_kernel<<<gblk, 256, 0, stream>>>(edge_feat, W_bf, e_ws, seg_sm,
                                             dst, out, e_cnt);
}
